// LSTMModel_32049045963274
// MI455X (gfx1250) — matrix operand order repaired and checked by reading, not compiled, not run on hardware
//
#include <hip/hip_runtime.h>
#include <hip/hip_bf16.h>

// Problem constants (match reference)
#define BB   64      // batch
#define SEQ  128     // encoder sequence length
#define EE   512     // embedding dim
#define HH   1024    // hidden dim
#define VV   10000   // vocab
#define TSTEPS 144   // 1 + 143 output steps
#define DSTEPS 143   // decode steps

typedef _Float16 f16;
typedef __attribute__((ext_vector_type(16))) _Float16 v16h;
typedef __attribute__((ext_vector_type(8)))  _Float16 v8h;
typedef __attribute__((ext_vector_type(8)))  float    v8f;

// ---------------------------------------------------------------------------
// WMMA fragment helpers (gfx1250 wave32 layouts, CDNA5 ISA §7.12.2)
//
// A (16x32 f16, M x K, row-major source, K-contiguous):
//   lane 0-15 : row M=lane,     elems K=[0..7] in v[0..7],  K=[16..23] in v[8..15]
//   lane 16-31: row M=lane-16,  elems K=[8..15],            K=[24..31]
//   => per-lane row pointer pre-offset by (lane>>4)*8; per k-iter read
//      [k0 .. k0+7] and [k0+16 .. k0+23].
// B (32x16 f16, K x N; sourced from W row-major [N,K] => column n = W row n):
//   lane 0-15 : col N=lane,     K=[0..7] in v[0..7],  K=[16..23] in v[8..15]
//   lane 16-31: col N=lane-16,  K=[8..15],            K=[24..31]
//   => per-lane row pointer pre-offset by (lane>>4)*8; read [k0..k0+7] and [k0+16..k0+23].
// ---------------------------------------------------------------------------
__device__ __forceinline__ v16h frag_from_row_a(const f16* __restrict__ arow, int k0) {
  v8h lo = *reinterpret_cast<const v8h*>(arow + k0);
  v8h hi = *reinterpret_cast<const v8h*>(arow + k0 + 16);
  v16h out;
#pragma unroll
  for (int i = 0; i < 8; ++i) { out[i] = lo[i]; out[i + 8] = hi[i]; }
  return out;
}

__device__ __forceinline__ v16h frag_from_row_b(const f16* __restrict__ brow, int k0) {
  v8h lo = *reinterpret_cast<const v8h*>(brow + k0);
  v8h hi = *reinterpret_cast<const v8h*>(brow + k0 + 16);
  v16h out;
#pragma unroll
  for (int i = 0; i < 8; ++i) { out[i] = lo[i]; out[i + 8] = hi[i]; }
  return out;
}

__device__ __forceinline__ float sigmoidf_(float x) {
  return 1.0f / (1.0f + expf(-x));
}

// ---------------------------------------------------------------------------
// Fused LSTM layer step: gates = x@Wih^T + h@Whh^T + bsum ; c,h update.
// Grid: HH/16 = 64 blocks; block: 128 threads (4 waves, one 16-row M-tile each).
// Each block owns hidden columns [n0, n0+16) and computes all 4 gate blocks
// (n0, n0+H, n0+2H, n0+3H) itself, so the pointwise LSTM cell fuses in-kernel.
//
// GATHER=true (layer 0): A rows are gathered directly from the f16 embedding
// table via per-row token ids — the embed kernel is fused away.
// GATHER=false (layer 1): A rows come from the dense layer-0 output.
// ---------------------------------------------------------------------------
template <bool GATHER>
__global__ void lstm_layer_kernel(const f16* __restrict__ xin, int kin,
                                  const int* __restrict__ tokens, int tstride, int toff,
                                  const f16* __restrict__ embf,  // [V, E] f16
                                  const f16* __restrict__ hprev, // [B, H] f16
                                  const f16* __restrict__ Wih,   // [4H, kin] f16
                                  const f16* __restrict__ Whh,   // [4H, H]  f16
                                  const float* __restrict__ bsum,// [4H]
                                  float* __restrict__ cstate,    // [B, H] in/out
                                  f16*   __restrict__ hout,      // [B, H]
                                  float* __restrict__ hout32) {  // [B, H] or null
  const int lane = threadIdx.x & 31;
  const int wave = threadIdx.x >> 5;
  const int m0 = wave << 4;                 // M-tile (batch rows)
  const int n0 = blockIdx.x << 4;           // hidden columns
  const int arowIdx = m0 + (lane & 15);     // this lane's A row (batch index)
  const int aoff = (lane >> 4) << 3;        // +8 elems for upper half-wave
  const int boff = (lane >> 4) << 3;        // +8 elems for upper half-wave

  // Hoisted per-lane row base pointers (K-loop becomes pure offset addressing).
  const f16* arow0;                         // layer input row
  if (GATHER) {
    const int t = tokens[arowIdx * tstride + toff];
    arow0 = embf + (size_t)t * EE + aoff;
  } else {
    arow0 = xin + (size_t)arowIdx * kin + aoff;
  }
  const f16* arowH = hprev + (size_t)arowIdx * HH + aoff;

  const f16* bih[4], *bhh[4];
#pragma unroll
  for (int g = 0; g < 4; ++g) {
    const int wrow = g * HH + n0 + (lane & 15);
    bih[g] = Wih + (size_t)wrow * kin + boff;
    bhh[g] = Whh + (size_t)wrow * HH + boff;
  }

  v8f acc[4] = {};                          // i, f, g, o accumulators

  // input GEMM: K = kin
  for (int k0 = 0; k0 < kin; k0 += 32) {
    v16h a = frag_from_row_a(arow0, k0);
#pragma unroll
    for (int g = 0; g < 4; ++g) {
      v16h b = frag_from_row_b(bih[g], k0);
      acc[g] = __builtin_amdgcn_wmma_f32_16x16x32_f16(
          false, a, false, b, (short)0, acc[g], false, false);
    }
  }
  // recurrent GEMM: K = H
  for (int k0 = 0; k0 < HH; k0 += 32) {
    v16h a = frag_from_row_a(arowH, k0);
#pragma unroll
    for (int g = 0; g < 4; ++g) {
      v16h b = frag_from_row_b(bhh[g], k0);
      acc[g] = __builtin_amdgcn_wmma_f32_16x16x32_f16(
          false, a, false, b, (short)0, acc[g], false, false);
    }
  }

  // Pointwise LSTM cell. D layout: lane 0-15 -> N=lane, M=r;
  //                      lane 16-31 -> N=lane-16, M=8+r.
  const int n = n0 + (lane & 15);
  const int mbase = m0 + ((lane >> 4) << 3);
  const float bi = bsum[n];
  const float bf = bsum[n + HH];
  const float bg = bsum[n + 2 * HH];
  const float bo = bsum[n + 3 * HH];
#pragma unroll
  for (int r = 0; r < 8; ++r) {
    const int m = mbase + r;
    const float iv = acc[0][r] + bi;
    const float fv = acc[1][r] + bf;
    const float gv = acc[2][r] + bg;
    const float ov = acc[3][r] + bo;
    const size_t idx = (size_t)m * HH + n;
    const float cp = cstate[idx];
    const float cn = sigmoidf_(fv) * cp + sigmoidf_(iv) * tanhf(gv);
    cstate[idx] = cn;
    const float hn = sigmoidf_(ov) * tanhf(cn);
    hout[idx] = (f16)hn;
    if (hout32) hout32[idx] = hn;
  }
}

// ---------------------------------------------------------------------------
// Logits GEMM: out[b, step, :] = h1 @ Wfc^T + bfc.  V = 10000 = 625 * 16.
// Grid: 625 blocks; block 128 threads (4 waves over M).
// ---------------------------------------------------------------------------
__global__ void logits_kernel(const f16* __restrict__ h1,     // [B, H]
                              const f16* __restrict__ Wfc,    // [V, H] f16
                              const float* __restrict__ bfc,  // [V]
                              float* __restrict__ out,        // [B, TSTEPS, V]
                              int step) {
  const int lane = threadIdx.x & 31;
  const int wave = threadIdx.x >> 5;
  const int m0 = wave << 4;
  const int n0 = blockIdx.x << 4;

  const f16* arow = h1 + (size_t)(m0 + (lane & 15)) * HH + (((lane >> 4) << 3));
  const f16* brow = Wfc + (size_t)(n0 + (lane & 15)) * HH + (((lane >> 4) << 3));

  v8f acc = {};
  for (int k0 = 0; k0 < HH; k0 += 32) {
    v16h a = frag_from_row_a(arow, k0);
    v16h b = frag_from_row_b(brow, k0);
    acc = __builtin_amdgcn_wmma_f32_16x16x32_f16(
        false, a, false, b, (short)0, acc, false, false);
  }

  const int n = n0 + (lane & 15);
  const int mbase = m0 + ((lane >> 4) << 3);
  const float bias = bfc[n];
#pragma unroll
  for (int r = 0; r < 8; ++r) {
    const int m = mbase + r;
    out[((size_t)m * TSTEPS + step) * VV + n] = acc[r] + bias;
  }
}

// ---------------------------------------------------------------------------
// Row argmax (first-occurrence-of-max semantics, matching jnp.argmax).
// One block per batch row; row b starts at base + b*rowStride.
// ---------------------------------------------------------------------------
__global__ void argmax_rows_kernel(const float* __restrict__ base,
                                   long long rowStride, int n,
                                   int* __restrict__ tok) {
  __shared__ float sval[256];
  __shared__ int   sidx[256];
  const float* row = base + (size_t)blockIdx.x * rowStride;
  const int tid = threadIdx.x;
  float best = -3.402823466e38f;
  int bidx = 0x7fffffff;
  for (int i = tid; i < n; i += blockDim.x) {
    const float v = row[i];
    if (v > best) { best = v; bidx = i; }
  }
  sval[tid] = best; sidx[tid] = bidx;
  __syncthreads();
  for (int s = 128; s > 0; s >>= 1) {
    if (tid < s) {
      const float v2 = sval[tid + s];
      const int   i2 = sidx[tid + s];
      if (v2 > sval[tid] || (v2 == sval[tid] && i2 < sidx[tid])) {
        sval[tid] = v2; sidx[tid] = i2;
      }
    }
    __syncthreads();
  }
  if (tid == 0) tok[blockIdx.x] = sidx[0];
}

__global__ void f32_to_f16_kernel(const float* __restrict__ src,
                                  f16* __restrict__ dst, int n) {
  const int i = blockIdx.x * blockDim.x + threadIdx.x;
  if (i < n) dst[i] = (f16)src[i];
}

__global__ void bias_sum_kernel(const float* __restrict__ a,
                                const float* __restrict__ b,
                                float* __restrict__ o, int n) {
  const int i = blockIdx.x * blockDim.x + threadIdx.x;
  if (i < n) o[i] = a[i] + b[i];
}

// ---------------------------------------------------------------------------
// Host-side orchestration (graph-capturable: only async launches/memsets).
// d_in order: x, emb, Wih0, Whh0, bih0, bhh0, Wih1, Whh1, bih1, bhh1, Wfc, bfc
// d_out: [B, 144, V] float.
// Workspace use: ~61 MB (f16 weight + embedding copies stay L2-resident).
// ---------------------------------------------------------------------------
extern "C" void kernel_launch(void* const* d_in, const int* in_sizes, int n_in,
                              void* d_out, int out_size, void* d_ws, size_t ws_size,
                              hipStream_t stream) {
  (void)in_sizes; (void)n_in; (void)out_size; (void)ws_size;
  const int*   x    = (const int*)  d_in[0];
  const float* emb  = (const float*)d_in[1];
  const float* Wih0 = (const float*)d_in[2];
  const float* Whh0 = (const float*)d_in[3];
  const float* bih0 = (const float*)d_in[4];
  const float* bhh0 = (const float*)d_in[5];
  const float* Wih1 = (const float*)d_in[6];
  const float* Whh1 = (const float*)d_in[7];
  const float* bih1 = (const float*)d_in[8];
  const float* bhh1 = (const float*)d_in[9];
  const float* Wfc  = (const float*)d_in[10];
  const float* bfc  = (const float*)d_in[11];
  float* out = (float*)d_out;

  // Workspace layout
  char* ws = (char*)d_ws;
  size_t off = 0;
  auto take = [&](size_t bytes) -> char* {
    char* p = ws + off;
    off = (off + bytes + 255) & ~(size_t)255;
    return p;
  };
  f16*   embf  = (f16*)take((size_t)VV * EE * sizeof(f16));
  f16*   Wih0h = (f16*)take((size_t)4 * HH * EE * sizeof(f16));
  f16*   Whh0h = (f16*)take((size_t)4 * HH * HH * sizeof(f16));
  f16*   Wih1h = (f16*)take((size_t)4 * HH * HH * sizeof(f16));
  f16*   Whh1h = (f16*)take((size_t)4 * HH * HH * sizeof(f16));
  f16*   Wfch  = (f16*)take((size_t)VV * HH * sizeof(f16));
  float* b0    = (float*)take(4 * HH * sizeof(float));
  float* b1    = (float*)take(4 * HH * sizeof(float));
  f16*   h0b[2], *h1b[2];
  h0b[0] = (f16*)take((size_t)BB * HH * sizeof(f16));
  h0b[1] = (f16*)take((size_t)BB * HH * sizeof(f16));
  h1b[0] = (f16*)take((size_t)BB * HH * sizeof(f16));
  h1b[1] = (f16*)take((size_t)BB * HH * sizeof(f16));
  float* c0  = (float*)take((size_t)BB * HH * sizeof(float));
  float* c1  = (float*)take((size_t)BB * HH * sizeof(float));
  float* h1f = (float*)take((size_t)BB * HH * sizeof(float));
  int*   tok = (int*)take(BB * sizeof(int));

  const int TPB = 256;
  auto cvt = [&](const float* s, f16* d, size_t n) {
    f32_to_f16_kernel<<<dim3((unsigned)((n + TPB - 1) / TPB)), TPB, 0, stream>>>(s, d, (int)n);
  };
  // 1) embedding + weights -> f16, bias sums
  cvt(emb,  embf,  (size_t)VV * EE);
  cvt(Wih0, Wih0h, (size_t)4 * HH * EE);
  cvt(Whh0, Whh0h, (size_t)4 * HH * HH);
  cvt(Wih1, Wih1h, (size_t)4 * HH * HH);
  cvt(Whh1, Whh1h, (size_t)4 * HH * HH);
  cvt(Wfc,  Wfch,  (size_t)VV * HH);
  bias_sum_kernel<<<(4 * HH + TPB - 1) / TPB, TPB, 0, stream>>>(bih0, bhh0, b0, 4 * HH);
  bias_sum_kernel<<<(4 * HH + TPB - 1) / TPB, TPB, 0, stream>>>(bih1, bhh1, b1, 4 * HH);

  // 2) zero initial state (every call: ws is not re-initialized by harness)
  hipMemsetAsync(h0b[0], 0, (size_t)BB * HH * sizeof(f16),   stream);
  hipMemsetAsync(h1b[0], 0, (size_t)BB * HH * sizeof(f16),   stream);
  hipMemsetAsync(c0,     0, (size_t)BB * HH * sizeof(float), stream);
  hipMemsetAsync(c1,     0, (size_t)BB * HH * sizeof(float), stream);

  const dim3 lstmGrid(HH / 16), lstmBlk(128);
  const dim3 logitGrid(VV / 16), logitBlk(128);

  // 3) encoder: layer 0 gathers its input rows straight from the embedding
  int p = 0;
  for (int t = 0; t < SEQ; ++t) {
    lstm_layer_kernel<true><<<lstmGrid, lstmBlk, 0, stream>>>(
        nullptr, EE, x, SEQ, t, embf,
        h0b[p], Wih0h, Whh0h, b0, c0, h0b[1 - p], nullptr);
    lstm_layer_kernel<false><<<lstmGrid, lstmBlk, 0, stream>>>(
        h0b[1 - p], HH, nullptr, 0, 0, nullptr,
        h1b[p], Wih1h, Whh1h, b1, c1, h1b[1 - p], h1f);
    p ^= 1;
  }

  // 4) first logits + quirk token (argmax over raw fp32 hidden state)
  logits_kernel<<<logitGrid, logitBlk, 0, stream>>>(h1b[p], Wfch, bfc, out, 0);
  argmax_rows_kernel<<<BB, 256, 0, stream>>>(h1f, (long long)HH, HH, tok);

  // 5) greedy decode (layer 0 gathers emb[tok] directly)
  for (int s = 1; s <= DSTEPS; ++s) {
    lstm_layer_kernel<true><<<lstmGrid, lstmBlk, 0, stream>>>(
        nullptr, EE, tok, 1, 0, embf,
        h0b[p], Wih0h, Whh0h, b0, c0, h0b[1 - p], nullptr);
    lstm_layer_kernel<false><<<lstmGrid, lstmBlk, 0, stream>>>(
        h0b[1 - p], HH, nullptr, 0, 0, nullptr,
        h1b[p], Wih1h, Whh1h, b1, c1, h1b[1 - p], nullptr);
    p ^= 1;
    logits_kernel<<<logitGrid, logitBlk, 0, stream>>>(h1b[p], Wfch, bfc, out, s);
    if (s < DSTEPS) {
      argmax_rows_kernel<<<BB, 256, 0, stream>>>(
          out + (size_t)s * VV, (long long)TSTEPS * VV, VV, tok);
    }
  }
}
